// TreeLSTM_23390391894198
// MI455X (gfx1250) — compile-verified, matrix operand order checked by reading, not run on hardware
//
#include <hip/hip_runtime.h>
#include <hip/hip_bf16.h>

typedef unsigned short u16;
typedef __attribute__((ext_vector_type(16))) __bf16 v16bf;
typedef __attribute__((ext_vector_type(8)))  float  v8f;
typedef __attribute__((ext_vector_type(4)))  unsigned int uint32x4;
typedef __attribute__((ext_vector_type(8)))  int          int32x8;
typedef __attribute__((ext_vector_type(4)))  int          int32x4;

#define LSTM_B  64
#define LSTM_S  512
#define LSTM_D  1024
#define LSTM_H  1024
#define LSTM_GH 4096   // 4 gates * H
#define NWG_SCAN 64    // one workgroup per 16-wide h-slab; must be co-resident

__device__ __forceinline__ u16 f32_to_bf16_rne(float f) {
    unsigned u = __float_as_uint(f);
    u += 0x7fffu + ((u >> 16) & 1u);
    return (u16)(u >> 16);
}

__device__ __forceinline__ v8f wmma_bf16(v16bf a, v16bf b, v8f c) {
    // D(f32 16x16) = A(bf16 16x32) * B(bf16 32x16) + C
    return __builtin_amdgcn_wmma_f32_16x16x32_bf16(
        false, a, false, b, (short)0, c, false, false);
}

#if __has_builtin(__builtin_amdgcn_tensor_load_to_lds)
#define HAVE_TDM 1
// 2D tile load via Tensor Data Mover (CDNA5 ISA sec.8).
// clang-23 / therock headers: 6-arg form (g0 x4, g1 x8, g2 x4, g3 x4, extra x8, cpol).
// lds_off: byte offset into this workgroup's LDS; gptr: tile start in global.
// Sizes in elements of 2 bytes (data_size code 1).
__device__ __forceinline__ void tdm_load_2d_bf16(unsigned lds_off, const void* gptr,
                                                 unsigned tile_rows, unsigned tile_cols,
                                                 unsigned tensor_d0, unsigned tensor_d1,
                                                 unsigned long long d0_stride_elems) {
    unsigned long long ga = (unsigned long long)gptr;
    uint32x4 g0;
    g0[0] = 1u;                                           // count=1, user descriptor
    g0[1] = lds_off;                                      // lds_addr (bytes)
    g0[2] = (unsigned)(ga & 0xffffffffu);                 // global_addr[31:0]
    g0[3] = (unsigned)((ga >> 32) & 0x1ffffffu) | (2u << 30); // addr[56:32] | type=2
    int32x8 g1;
    g1[0] = (int)(1u << 16);                              // data_size=1 (2B), wg_mask=0
    g1[1] = (int)((tensor_d0 & 0xffffu) << 16);           // tensor_dim0[15:0]
    g1[2] = (int)(((tensor_d0 >> 16) & 0xffffu) | ((tensor_d1 & 0xffffu) << 16));
    g1[3] = (int)(((tensor_d1 >> 16) & 0xffffu) | ((tile_cols & 0xffffu) << 16)); // tile_dim0
    g1[4] = (int)(tile_rows & 0xffffu);                   // tile_dim1 (tile_dim2=0)
    g1[5] = (int)(unsigned)(d0_stride_elems & 0xffffffffu);       // dim0_stride[31:0]
    g1[6] = (int)(unsigned)((d0_stride_elems >> 32) & 0xffffu);   // dim0_stride[47:32]
    g1[7] = 0;
    int32x4 z4 = {0, 0, 0, 0};
    int32x8 z8 = {0, 0, 0, 0, 0, 0, 0, 0};
    __builtin_amdgcn_tensor_load_to_lds(g0, g1, z4, z4, z8, 0);
}
#else
#define HAVE_TDM 0
#endif

// ---------------------------------------------------------------------------
// fp32 -> bf16 conversion
// ---------------------------------------------------------------------------
__global__ __launch_bounds__(256) void cvt_f32_bf16(const float* __restrict__ s,
                                                    u16* __restrict__ d, int n) {
    int i = blockIdx.x * 256 + threadIdx.x;
    if (i < n) d[i] = f32_to_bf16_rne(s[i]);
}

// ---------------------------------------------------------------------------
// xp[s,b,gcol] = sum_k x[b,s,k] * Wx[gcol,k] + bx[gcol]
// M = S*B = 32768 (row r = s*64+b), N = 4096, K = 1024.
// Block 256 thr = 8 waves, tile 128x128, K-chunk 32, LDS double-buffered:
// global loads for chunk k+1 are in flight while WMMAs consume chunk k from LDS.
// ---------------------------------------------------------------------------
__global__ __launch_bounds__(256) void gemm_xp(const u16* __restrict__ xbf,
                                               const u16* __restrict__ wbf,
                                               const float* __restrict__ bx,
                                               float* __restrict__ xp) {
    __shared__ u16 sA[2][128 * 32];   // 8 KB per buffer
    __shared__ u16 sB[2][128 * 32];

    const int tid  = threadIdx.x;
    const int lane = tid & 31;
    const int wave = tid >> 5;
    const int lr = lane & 15;   // M (A tile) / N (B,C tiles)
    const int lh = lane >> 4;   // K-half selector
    const int rowBase = blockIdx.y * 128;
    const int colBase = blockIdx.x * 128;
    const int bufStride = 128 * 32;   // elements per LDS buffer

    // staging role: each thread copies one 32-byte half-row per array per chunk
    const int srow  = tid >> 1;       // 0..127
    const int shalf = tid & 1;
    const int ar = rowBase + srow;                 // global M row = s*64+b
    const int ab = ar & 63, as_ = ar >> 6;
    const u16* aSrc = xbf + ((size_t)ab * LSTM_S + as_) * LSTM_D + shalf * 16;
    const u16* bSrc = wbf + (size_t)(colBase + srow) * LSTM_D + shalf * 16;
    u16* aDst = &sA[0][srow * 32 + shalf * 16];
    u16* bDst = &sB[0][srow * 32 + shalf * 16];

    v8f acc[8];
#pragma unroll
    for (int tc = 0; tc < 8; ++tc) acc[tc] = (v8f){0.f,0.f,0.f,0.f,0.f,0.f,0.f,0.f};

    // prologue: chunk 0 -> buffer 0
    {
        uint4 ra0 = *(const uint4*)(aSrc);
        uint4 ra1 = *(const uint4*)(aSrc + 8);
        uint4 rb0 = *(const uint4*)(bSrc);
        uint4 rb1 = *(const uint4*)(bSrc + 8);
        *(uint4*)(aDst)     = ra0;  *(uint4*)(aDst + 8) = ra1;
        *(uint4*)(bDst)     = rb0;  *(uint4*)(bDst + 8) = rb1;
    }
    __syncthreads();

    const u16* aTile = &sA[0][(wave * 16 + lr) * 32 + lh * 16];
    const u16* bTile = &sB[0][lr * 32 + lh * 16];

    for (int kc = 0; kc < 31; ++kc) {
        const int cur = kc & 1;
        const int nxt = cur ^ 1;
        const int kkn = (kc + 1) * 32;
        // issue next chunk's global loads (overlap with WMMAs below)
        uint4 ra0 = *(const uint4*)(aSrc + kkn);
        uint4 ra1 = *(const uint4*)(aSrc + kkn + 8);
        uint4 rb0 = *(const uint4*)(bSrc + kkn);
        uint4 rb1 = *(const uint4*)(bSrc + kkn + 8);

        v16bf a = *(const v16bf*)(aTile + cur * bufStride);
#pragma unroll
        for (int tc = 0; tc < 8; ++tc) {
            v16bf bt = *(const v16bf*)(bTile + cur * bufStride + tc * 16 * 32);
            acc[tc] = wmma_bf16(a, bt, acc[tc]);
        }
        *(uint4*)(aDst + nxt * bufStride)     = ra0;
        *(uint4*)(aDst + nxt * bufStride + 8) = ra1;
        *(uint4*)(bDst + nxt * bufStride)     = rb0;
        *(uint4*)(bDst + nxt * bufStride + 8) = rb1;
        __syncthreads();
    }
    {   // final chunk (kc=31) lives in buffer 1
        v16bf a = *(const v16bf*)(aTile + bufStride);
#pragma unroll
        for (int tc = 0; tc < 8; ++tc) {
            v16bf bt = *(const v16bf*)(bTile + bufStride + tc * 16 * 32);
            acc[tc] = wmma_bf16(a, bt, acc[tc]);
        }
    }

#pragma unroll
    for (int tc = 0; tc < 8; ++tc) {
        const int col = colBase + tc * 16 + lr;
        const float bias = bx[col];
#pragma unroll
        for (int v = 0; v < 8; ++v) {
            const int m = rowBase + wave * 16 + v + 8 * lh;
            xp[(size_t)m * LSTM_GH + col] = acc[tc][v] + bias;
        }
    }
}

// ---------------------------------------------------------------------------
// Persistent LSTM scan. 64 workgroups; block js owns h-columns [16js,16js+16).
// Wh slab (64 outcols x 1024 K bf16 = 128KB) staged to LDS once via TDM.
// ---------------------------------------------------------------------------
__global__ __launch_bounds__(256) void lstm_scan(const float* __restrict__ xp,
                                                 const u16* __restrict__ whbf,
                                                 const float* __restrict__ bh,
                                                 u16* __restrict__ hbf,
                                                 unsigned* __restrict__ bar,
                                                 float* __restrict__ out) {
    extern __shared__ char smem[];
    u16*   sWh = (u16*)smem;                          // 64 x 1024 bf16 = 131072 B
    float* sG  = (float*)(smem + 131072);             // 64 x 64  f32  =  16384 B
    float* sC  = (float*)(smem + 131072 + 16384);     // 64 x 16  f32  =   4096 B

    const int js   = blockIdx.x;          // 0..63
    const int tid  = threadIdx.x;
    const int lane = tid & 31;
    const int wave = tid >> 5;
    const int g    = wave & 3;            // gate: 0=i 1=f 2=o 3=u
    const int rt0  = (wave >> 2) * 2;     // row tiles rt0, rt0+1 (16 rows each)
    const int lr   = lane & 15;
    const int lh   = lane >> 4;

    // --- stage Wh slab into LDS: slab row g'*16+n  <->  Wh[g', js*16+n, :] ---
#if HAVE_TDM
    if (wave == 0) {
        const unsigned ldsBase = __builtin_amdgcn_groupstaticsize();
#pragma unroll
        for (int gq = 0; gq < 4; ++gq) {
            const u16* src = whbf + ((size_t)gq * LSTM_H + js * 16) * LSTM_H;
            tdm_load_2d_bf16(ldsBase + gq * 16 * 2048, src,
                             /*tile rows*/16, /*tile cols(elems)*/LSTM_H,
                             /*tensor d0*/LSTM_H,
                             /*tensor d1*/(unsigned)(LSTM_GH - (gq * LSTM_H + js * 16)),
                             /*d0 stride*/LSTM_H);
        }
        __builtin_amdgcn_s_wait_tensorcnt(0);
    }
#else
    {
        const int q = tid >> 2;           // slab row 0..63
        const int part = tid & 3;         // 512-byte quarter of a 2048-byte row
        const int gq = q >> 4, nq = q & 15;
        const uint4* src = (const uint4*)(whbf + ((size_t)gq * LSTM_H + js * 16 + nq) * LSTM_H) + part * 32;
        uint4* dst = (uint4*)(sWh + (size_t)q * LSTM_H) + part * 32;
#pragma unroll
        for (int i = 0; i < 32; ++i) dst[i] = src[i];
    }
#endif
    for (int i = tid; i < 64 * 16; i += 256) sC[i] = 0.f;   // c0 = 0
    __syncthreads();

    const int colLocal = g * 16 + lr;                 // 0..63 within slab
    const int gcol = g * LSTM_H + js * 16 + lr;       // global gate column
    const float bhv = bh[gcol];
    const u16* brow = sWh + (size_t)colLocal * LSTM_H + lh * 16;

    for (int t = 0; t < LSTM_S; ++t) {
        // accumulators seeded with xp[t] + bh
        v8f acc0, acc1;
#pragma unroll
        for (int v = 0; v < 8; ++v) {
            const int m0 = rt0 * 16 + v + 8 * lh;
            acc0[v] = xp[((size_t)t * LSTM_B + m0)      * LSTM_GH + gcol] + bhv;
            acc1[v] = xp[((size_t)t * LSTM_B + m0 + 16) * LSTM_GH + gcol] + bhv;
        }
        const u16* hrow0 = hbf + (size_t)(rt0 * 16 + lr) * LSTM_H + lh * 16;
        const u16* hrow1 = hrow0 + 16 * LSTM_H;
        // software-pipelined K loop: A loads run one chunk ahead of the WMMAs
        v16bf a0 = *(const v16bf*)(hrow0);
        v16bf a1 = *(const v16bf*)(hrow1);
#pragma unroll 4
        for (int kk = 0; kk < LSTM_H - 32; kk += 32) {
            v16bf bt  = *(const v16bf*)(brow + kk);          // LDS
            v16bf na0 = *(const v16bf*)(hrow0 + kk + 32);    // global, prefetch
            v16bf na1 = *(const v16bf*)(hrow1 + kk + 32);
            acc0 = wmma_bf16(a0, bt, acc0);
            acc1 = wmma_bf16(a1, bt, acc1);
            a0 = na0; a1 = na1;
        }
        {
            v16bf bt = *(const v16bf*)(brow + LSTM_H - 32);
            acc0 = wmma_bf16(a0, bt, acc0);
            acc1 = wmma_bf16(a1, bt, acc1);
        }
        // exchange gate tiles through LDS
#pragma unroll
        for (int v = 0; v < 8; ++v) {
            const int m0 = rt0 * 16 + v + 8 * lh;
            sG[m0 * 64 + colLocal]        = acc0[v];
            sG[(m0 + 16) * 64 + colLocal] = acc1[v];
        }
        __syncthreads();

        // elementwise cell update: 64 rows x 16 cols, 4 per thread
#pragma unroll
        for (int e = 0; e < 4; ++e) {
            const int idx = tid + e * 256;            // = row*16 + j
            const int row = idx >> 4;
            const int j   = idx & 15;
            const float gi = sG[row * 64 +  0 + j];
            const float gf = sG[row * 64 + 16 + j];
            const float go = sG[row * 64 + 32 + j];
            const float gu = sG[row * 64 + 48 + j];
            const float i_ = 1.f / (1.f + __expf(-gi));
            const float f_ = 1.f / (1.f + __expf(-gf));
            const float o_ = 1.f / (1.f + __expf(-go));
            const float u_ = tanhf(gu);
            const float cn = f_ * sC[idx] + i_ * u_;
            const float hn = o_ * tanhf(cn);
            sC[idx] = cn;
            hbf[(size_t)row * LSTM_H + js * 16 + j] = f32_to_bf16_rne(hn);
            if (t == LSTM_S - 1) out[(size_t)row * LSTM_H + js * 16 + j] = hn;
        }
        __threadfence();      // make h stores agent-visible
        __syncthreads();      // also guards sG reuse next iteration

        if (t != LSTM_S - 1) {
            if (tid == 0) {
                __hip_atomic_fetch_add(&bar[t], 1u, __ATOMIC_RELEASE, __HIP_MEMORY_SCOPE_AGENT);
                unsigned v;
                do {
                    __builtin_amdgcn_s_sleep(1);
                    v = __hip_atomic_load(&bar[t], __ATOMIC_ACQUIRE, __HIP_MEMORY_SCOPE_AGENT);
                } while (v < (unsigned)NWG_SCAN);
            }
            __syncthreads();
            __builtin_amdgcn_fence(__ATOMIC_ACQUIRE, "agent");  // refresh WGP$ view of h
        }
    }
}

// ---------------------------------------------------------------------------
// host side
// ---------------------------------------------------------------------------
extern "C" void kernel_launch(void* const* d_in, const int* in_sizes, int n_in,
                              void* d_out, int out_size, void* d_ws, size_t ws_size,
                              hipStream_t stream) {
    const float* x  = (const float*)d_in[0];   // [B,S,D]
    const float* Wx = (const float*)d_in[1];   // [4,H,D]
    const float* bx = (const float*)d_in[2];   // [4,H]
    const float* Wh = (const float*)d_in[3];   // [4,H,H]
    const float* bh = (const float*)d_in[4];   // [4,H]
    float* out = (float*)d_out;                // [B,H]

    char* ws = (char*)d_ws;
    const size_t off_xp  = 0;                                              // S*B*4H f32
    const size_t off_xb  = off_xp + (size_t)LSTM_S * LSTM_B * LSTM_GH * 4; // B*S*D bf16
    const size_t off_wxb = off_xb + (size_t)LSTM_B * LSTM_S * LSTM_D * 2;  // 4H*D bf16
    const size_t off_whb = off_wxb + (size_t)LSTM_GH * LSTM_D * 2;         // 4H*H bf16
    const size_t off_h   = off_whb + (size_t)LSTM_GH * LSTM_H * 2;         // B*H bf16
    const size_t off_bar = off_h + (size_t)LSTM_B * LSTM_H * 2;            // S u32

    float* xp   = (float*)(ws + off_xp);
    u16*   xbf  = (u16*)(ws + off_xb);
    u16*   wxbf = (u16*)(ws + off_wxb);
    u16*   whbf = (u16*)(ws + off_whb);
    u16*   hbf  = (u16*)(ws + off_h);
    unsigned* bar = (unsigned*)(ws + off_bar);

    // zero h0 and the 512 one-shot barrier counters (graph-capturable)
    (void)hipMemsetAsync(ws + off_h, 0, (size_t)LSTM_B * LSTM_H * 2 + (size_t)LSTM_S * 4, stream);

    const int nx  = LSTM_B * LSTM_S * LSTM_D;
    const int nwx = LSTM_GH * LSTM_D;
    const int nwh = LSTM_GH * LSTM_H;
    cvt_f32_bf16<<<(nx  + 255) / 256, 256, 0, stream>>>(x,  xbf,  nx);
    cvt_f32_bf16<<<(nwx + 255) / 256, 256, 0, stream>>>(Wx, wxbf, nwx);
    cvt_f32_bf16<<<(nwh + 255) / 256, 256, 0, stream>>>(Wh, whbf, nwh);

    // xp = x @ Wx^T + bx : grid (N/128, M/128) = (32, 256)
    gemm_xp<<<dim3(LSTM_GH / 128, (LSTM_S * LSTM_B) / 128), 256, 0, stream>>>(xbf, wxbf, bx, xp);

    // persistent scan: 64 workgroups, 151.5 KB dynamic LDS each
    const size_t smem = 131072 + 16384 + 4096;
    lstm_scan<<<NWG_SCAN, 256, smem, stream>>>(xp, whbf, bh, hbf, bar, out);
}
